// HyperGraphStructuralLayer_Louvain_and_knn_19825569038843
// MI455X (gfx1250) — compile-verified, matrix operand order checked by reading, not run on hardware
//
#include <hip/hip_runtime.h>
#include <math.h>

typedef __attribute__((ext_vector_type(16))) _Float16 v16h;
typedef __attribute__((ext_vector_type(8)))  float    v8f;
typedef __attribute__((ext_vector_type(4)))  unsigned u32x4;

#define HDIM 128
#define NEG_SLOPE 0.2f

// ---------- helpers: order-preserving float<->uint for atomicMax-based segment max ----------
static __device__ __forceinline__ unsigned enc_f(float f) {
  unsigned u = __float_as_uint(f);
  return (u & 0x80000000u) ? ~u : (u | 0x80000000u);
}
static __device__ __forceinline__ float dec_f(unsigned u) {
  return (u & 0x80000000u) ? __uint_as_float(u ^ 0x80000000u) : __uint_as_float(~u);
}

// ---------- f32 -> f16 convert ----------
__global__ void k_f32_to_f16(const float* __restrict__ in, _Float16* __restrict__ out, int n) {
  int t = blockIdx.x * blockDim.x + threadIdx.x;
  if (t < n) out[t] = (_Float16)in[t];
}

// ---------- pack 128x128 W (f32) into WMMA B-fragment order (f16, packed u32) ----------
// Layout: Bp[((tn*4 + s)*32 + lane)*8 + p]  -> per-lane 8 consecutive dwords = one fragment
__global__ void k_pack_w(const float* __restrict__ W, unsigned* __restrict__ Bp) {
  int tid = blockIdx.x * blockDim.x + threadIdx.x;
  if (tid >= 8 * 4 * 32 * 8) return;
  int p    = tid & 7;
  int lane = (tid >> 3) & 31;
  int s    = (tid >> 8) & 3;
  int tn   = tid >> 10;
  int h    = lane >> 4;
  int nn   = tn * 16 + (lane & 15);
  int kb   = 32 * s + 8 * h + ((p < 4) ? (2 * p) : (16 + 2 * (p - 4)));
  union { _Float16 f[2]; unsigned u; } pk;
  pk.f[0] = (_Float16)W[kb * HDIM + nn];
  pk.f[1] = (_Float16)W[(kb + 1) * HDIM + nn];
  Bp[tid] = pk.u;
}

// ---------- C[N,128] = A[N,128] @ W[128,128] via v_wmma_f32_16x16x32_f16 ----------
// One wave per 16-row M tile; 8 N-tiles held in 8 v8f accumulators; K=128 in 4 steps.
__global__ void k_gemm_wmma(const _Float16* __restrict__ A, const unsigned* __restrict__ Bp,
                            float* __restrict__ C, int nrows) {
  int wid  = (blockIdx.x * blockDim.x + threadIdx.x) >> 5;
  int lane = threadIdx.x & 31;
  if (wid * 16 >= nrows) return;
  int m = wid * 16 + (lane & 15);
  int h = lane >> 4;
  v8f acc[8] = {};
#pragma unroll
  for (int s = 0; s < 4; ++s) {
    // A fragment (16-bit A 16x32 layout): lane holds K = [32s+8h .. +7] and [32s+16+8h .. +7]
    union { u32x4 u[2]; v16h v; } af;
    const u32x4* ap = reinterpret_cast<const u32x4*>(A + (size_t)m * HDIM + 32 * s + 8 * h);
    af.u[0] = ap[0];
    af.u[1] = ap[2];
#pragma unroll
    for (int tn = 0; tn < 8; ++tn) {
      union { u32x4 u[2]; v16h v; } bf;
      const u32x4* bp = reinterpret_cast<const u32x4*>(Bp + (((tn * 4 + s) * 32 + lane) << 3));
      bf.u[0] = bp[0];
      bf.u[1] = bp[1];
      acc[tn] = __builtin_amdgcn_wmma_f32_16x16x32_f16(
          false, af.v, false, bf.v, (short)0, acc[tn], false, false);
    }
  }
#pragma unroll
  for (int tn = 0; tn < 8; ++tn) {
#pragma unroll
    for (int i = 0; i < 8; ++i) {
      int rr = wid * 16 + i + 8 * h;      // C layout: VGPR i -> row i (+8 for upper half-wave)
      int cc = tn * 16 + (lane & 15);
      C[(size_t)rr * HDIM + cc] = acc[tn][i];
    }
  }
}

// ---------- per-node attention dots: si = xw . att[0:128], sj = xw . att[128:256] ----------
__global__ void k_node_att(const float* __restrict__ xw, const float* __restrict__ att,
                           float* __restrict__ si, float* __restrict__ sj, int nN) {
  int wid  = (blockIdx.x * blockDim.x + threadIdx.x) >> 5;
  int lane = threadIdx.x & 31;
  if (wid >= nN) return;
  const float4* xp = reinterpret_cast<const float4*>(xw + (size_t)wid * HDIM);
  const float4* ai = reinterpret_cast<const float4*>(att);
  const float4* aj = reinterpret_cast<const float4*>(att + HDIM);
  float4 xv = xp[lane];
  float4 av = ai[lane];
  float4 bv = aj[lane];
  float pi = xv.x * av.x + xv.y * av.y + xv.z * av.z + xv.w * av.w;
  float pj = xv.x * bv.x + xv.y * bv.y + xv.z * bv.z + xv.w * bv.w;
#pragma unroll
  for (int off = 16; off > 0; off >>= 1) {
    pi += __shfl_xor(pi, off, 32);
    pj += __shfl_xor(pj, off, 32);
  }
  if (lane == 0) { si[wid] = pi; sj[wid] = pj; }
}

// ---------- incidence degree counts (layer-invariant) ----------
__global__ void k_counts(const long long* __restrict__ hidx, unsigned* __restrict__ Dcnt,
                         unsigned* __restrict__ Bcnt, int E) {
  int e = blockIdx.x * blockDim.x + threadIdx.x;
  if (e >= E) return;
  int r = (int)hidx[e];
  int c = (int)hidx[(size_t)E + e];
  atomicAdd(&Dcnt[r], 1u);
  atomicAdd(&Bcnt[c], 1u);
}

// ---------- segment max of leaky-relu logits over col ----------
__global__ void k_edge_max(const long long* __restrict__ hidx, const float* __restrict__ si,
                           const float* __restrict__ sj, unsigned* __restrict__ mu, int E) {
  int e = blockIdx.x * blockDim.x + threadIdx.x;
  if (e >= E) return;
  int r = (int)hidx[e];
  int c = (int)hidx[(size_t)E + e];
  float l = si[r] + sj[c];
  l = (l >= 0.f) ? l : NEG_SLOPE * l;
  atomicMax(&mu[c], enc_f(l));
}

// ---------- exp(logit - max) and segment sum over col ----------
__global__ void k_edge_exp(const long long* __restrict__ hidx, const float* __restrict__ si,
                           const float* __restrict__ sj, const unsigned* __restrict__ mu,
                           float* __restrict__ alpha, float* __restrict__ ssum, int E) {
  int e = blockIdx.x * blockDim.x + threadIdx.x;
  if (e >= E) return;
  int r = (int)hidx[e];
  int c = (int)hidx[(size_t)E + e];
  float l = si[r] + sj[c];
  l = (l >= 0.f) ? l : NEG_SLOPE * l;
  float ev = __expf(l - dec_f(mu[c]));
  alpha[e] = ev;
  atomicAdd(&ssum[c], ev);
}

// ---------- edge_feat_raw[col] += alpha * xw[row]  (wave per edge, float4/lane) ----------
__global__ void k_scatter1(const long long* __restrict__ hidx, const float* __restrict__ alpha,
                           const float* __restrict__ ssum, const float* __restrict__ xw,
                           float* __restrict__ ef, int E) {
  int wid  = (blockIdx.x * blockDim.x + threadIdx.x) >> 5;
  int lane = threadIdx.x & 31;
  if (wid >= E) return;
  int r = (int)hidx[wid];
  int c = (int)hidx[(size_t)E + wid];
  float a = alpha[wid] / (ssum[c] + 1e-16f);
  float4 v = reinterpret_cast<const float4*>(xw + (size_t)r * HDIM)[lane];
  float* dst = ef + (size_t)c * HDIM + lane * 4;
  atomicAdd(dst + 0, a * v.x);
  atomicAdd(dst + 1, a * v.y);
  atomicAdd(dst + 2, a * v.z);
  atomicAdd(dst + 3, a * v.w);
}

// ---------- out_raw[row] += alpha * B_inv[col] * edge_feat_raw[col] ----------
__global__ void k_scatter2(const long long* __restrict__ hidx, const float* __restrict__ alpha,
                           const float* __restrict__ ssum, const unsigned* __restrict__ Bcnt,
                           const float* __restrict__ ef, float* __restrict__ oraw, int E) {
  int wid  = (blockIdx.x * blockDim.x + threadIdx.x) >> 5;
  int lane = threadIdx.x & 31;
  if (wid >= E) return;
  int r = (int)hidx[wid];
  int c = (int)hidx[(size_t)E + wid];
  unsigned bc = Bcnt[c];
  float binv = bc ? 1.0f / (float)bc : 0.0f;
  float a = binv * alpha[wid] / (ssum[c] + 1e-16f);
  float4 v = reinterpret_cast<const float4*>(ef + (size_t)c * HDIM)[lane];
  float* dst = oraw + (size_t)r * HDIM + lane * 4;
  atomicAdd(dst + 0, a * v.x);
  atomicAdd(dst + 1, a * v.y);
  atomicAdd(dst + 2, a * v.z);
  atomicAdd(dst + 3, a * v.w);
}

// ---------- layer-1 finalize: h1 = prelu(D_inv*oraw + b) -> written directly as f16 ----------
__global__ void k_finalize1(const float* __restrict__ oraw, const unsigned* __restrict__ Dcnt,
                            const float* __restrict__ b, const float* __restrict__ pa,
                            _Float16* __restrict__ xf16, int total) {
  int t = blockIdx.x * blockDim.x + threadIdx.x;
  if (t >= total) return;
  int n = t >> 7, hh = t & 127;
  unsigned d = Dcnt[n];
  float dinv = d ? 1.0f / (float)d : 0.0f;
  float v = dinv * oraw[t] + b[hh];
  float a = pa[0];
  v = (v >= 0.f) ? v : a * v;
  xf16[t] = (_Float16)v;
}

// ---------- layer-2 finalize: out = prelu(D_inv*oraw + b + x) ----------
__global__ void k_finalize2(const float* __restrict__ oraw, const unsigned* __restrict__ Dcnt,
                            const float* __restrict__ b, const float* __restrict__ x,
                            const float* __restrict__ pa, float* __restrict__ out, int total) {
  int t = blockIdx.x * blockDim.x + threadIdx.x;
  if (t >= total) return;
  int n = t >> 7, hh = t & 127;
  unsigned d = Dcnt[n];
  float dinv = d ? 1.0f / (float)d : 0.0f;
  float v = dinv * oraw[t] + b[hh] + x[t];
  float a = pa[0];
  out[t] = (v >= 0.f) ? v : a * v;
}

extern "C" void kernel_launch(void* const* d_in, const int* in_sizes, int n_in,
                              void* d_out, int out_size, void* d_ws, size_t ws_size,
                              hipStream_t stream) {
  const float*     x    = (const float*)d_in[0];
  const long long* hidx = (const long long*)d_in[1];
  const float*     W1   = (const float*)d_in[2];
  const float*     att1 = (const float*)d_in[3];
  const float*     b1   = (const float*)d_in[4];
  const float*     W2   = (const float*)d_in[5];
  const float*     att2 = (const float*)d_in[6];
  const float*     b2   = (const float*)d_in[7];
  const float*     pa   = (const float*)d_in[8];
  float* out = (float*)d_out;

  const int N = in_sizes[0] / HDIM;
  const int E = in_sizes[1] / 2;
  const int NH = N * HDIM;

  // ---- carve workspace ----
  char* ws = (char*)d_ws;
  size_t off = 0;
  auto carve = [&](size_t bytes) -> void* {
    off = (off + 255) & ~(size_t)255;
    void* p = ws + off;
    off += bytes;
    return p;
  };
  _Float16* xf16  = (_Float16*)carve((size_t)NH * 2);
  unsigned* Wpack = (unsigned*)carve(8 * 4 * 32 * 8 * 4);
  float*    xw    = (float*)carve((size_t)NH * 4);
  float*    si    = (float*)carve((size_t)N * 4);
  float*    sj    = (float*)carve((size_t)N * 4);
  unsigned* mu    = (unsigned*)carve((size_t)N * 4);
  float*    ssum  = (float*)carve((size_t)N * 4);
  float*    alpha = (float*)carve((size_t)E * 4);
  unsigned* Dcnt  = (unsigned*)carve((size_t)N * 4);
  unsigned* Bcnt  = (unsigned*)carve((size_t)N * 4);
  float*    ef    = (float*)carve((size_t)NH * 4);
  float*    oraw  = (float*)carve((size_t)NH * 4);

  const int TPB = 256;
  int gNH    = (NH + TPB - 1) / TPB;
  int gE     = (E + TPB - 1) / TPB;
  int gEwave = (E + 7) / 8;          // wave-per-edge, 8 waves/block
  int gNwave = (N + 7) / 8;          // wave-per-node
  int mTiles = (N + 15) / 16;
  int gGemm  = (mTiles + 7) / 8;     // 8 waves/block (256 threads)

  // ---- degree counts (shared by both layers) ----
  hipMemsetAsync(Dcnt, 0, (size_t)N * 4, stream);
  hipMemsetAsync(Bcnt, 0, (size_t)N * 4, stream);
  k_counts<<<gE, TPB, 0, stream>>>(hidx, Dcnt, Bcnt, E);

  // =================== layer 1 ===================
  k_f32_to_f16<<<gNH, TPB, 0, stream>>>(x, xf16, NH);
  k_pack_w<<<(8192 + TPB - 1) / TPB, TPB, 0, stream>>>(W1, Wpack);
  k_gemm_wmma<<<gGemm, TPB, 0, stream>>>(xf16, Wpack, xw, N);
  k_node_att<<<gNwave, TPB, 0, stream>>>(xw, att1, si, sj, N);

  hipMemsetAsync(mu, 0, (size_t)N * 4, stream);
  k_edge_max<<<gE, TPB, 0, stream>>>(hidx, si, sj, mu, E);
  hipMemsetAsync(ssum, 0, (size_t)N * 4, stream);
  k_edge_exp<<<gE, TPB, 0, stream>>>(hidx, si, sj, mu, alpha, ssum, E);

  hipMemsetAsync(ef, 0, (size_t)NH * 4, stream);
  k_scatter1<<<gEwave, TPB, 0, stream>>>(hidx, alpha, ssum, xw, ef, E);
  hipMemsetAsync(oraw, 0, (size_t)NH * 4, stream);
  k_scatter2<<<gEwave, TPB, 0, stream>>>(hidx, alpha, ssum, Bcnt, ef, oraw, E);
  k_finalize1<<<gNH, TPB, 0, stream>>>(oraw, Dcnt, b1, pa, xf16, NH);  // h1 -> f16

  // =================== layer 2 ===================
  k_pack_w<<<(8192 + TPB - 1) / TPB, TPB, 0, stream>>>(W2, Wpack);
  k_gemm_wmma<<<gGemm, TPB, 0, stream>>>(xf16, Wpack, xw, N);
  k_node_att<<<gNwave, TPB, 0, stream>>>(xw, att2, si, sj, N);

  hipMemsetAsync(mu, 0, (size_t)N * 4, stream);
  k_edge_max<<<gE, TPB, 0, stream>>>(hidx, si, sj, mu, E);
  hipMemsetAsync(ssum, 0, (size_t)N * 4, stream);
  k_edge_exp<<<gE, TPB, 0, stream>>>(hidx, si, sj, mu, alpha, ssum, E);

  hipMemsetAsync(ef, 0, (size_t)NH * 4, stream);
  k_scatter1<<<gEwave, TPB, 0, stream>>>(hidx, alpha, ssum, xw, ef, E);
  hipMemsetAsync(oraw, 0, (size_t)NH * 4, stream);
  k_scatter2<<<gEwave, TPB, 0, stream>>>(hidx, alpha, ssum, Bcnt, ef, oraw, E);
  k_finalize2<<<gNH, TPB, 0, stream>>>(oraw, Dcnt, b2, x, pa, out, NH);
}
